// LayeredRNN_18107582120192
// MI455X (gfx1250) — matrix operand order repaired and checked by reading, not compiled, not run on hardware
//
#include <hip/hip_runtime.h>
#include <hip/hip_bf16.h>
#include <stdint.h>

// Problem constants (from reference): B=64, T=2048, N_in=128, N_h=512, N_out=128
#define RNN_B    64
#define RNN_T    2048
#define RNN_NIN  128
#define RNN_NH   512
#define RNN_NOUT 128
#define RNN_ALPHA 0.1f

typedef __attribute__((ext_vector_type(16))) __bf16 v16bf;
typedef __attribute__((ext_vector_type(8)))  __bf16 v8bf;
typedef __attribute__((ext_vector_type(8)))  float  v8f;
typedef __attribute__((ext_vector_type(4)))  unsigned int v4u;
typedef __attribute__((ext_vector_type(8)))  int v8i;
typedef __attribute__((ext_vector_type(4)))  int v4i;

// Workspace layout (bf16, pre-transposed weights: W^T stored N-major, stride K)
#define WS_WHHT 0u                                   // [NH ][NH ]  512 KB
#define WS_WUHT (RNN_NH * RNN_NH * 2u)               // [NH ][NIN] 128 KB
#define WS_WHYT (WS_WUHT + RNN_NH * RNN_NIN * 2u)    // [NOUT][NH] 128 KB

// Dynamic LDS layout (no static __shared__, so dynamic base offset == 0)
#define LDS_HS     0u      // float  hs[16][NH]   32 KB   fp32 hidden state
#define LDS_USTAGE 32768u  // float  us[16][NIN]   8 KB   TDM staging of u_t
#define LDS_HB     40960u  // __bf16 hb[16][NH]   16 KB   bf16 hidden (WMMA A)
#define LDS_UB     57344u  // __bf16 ub[16][NIN]   4 KB   bf16 u_t    (WMMA A)
#define LDS_BYTES  61440u

static __device__ inline v8f wmma_bf16(v16bf A, v16bf B, v8f C) {
    // D = A(16x32 bf16) x B(32x16 bf16) + C(16x16 f32)
    return __builtin_amdgcn_wmma_f32_16x16x32_bf16(
        /*neg_a=*/false, A, /*neg_b=*/false, B,
        /*c_mod=*/(short)0, C, /*reuse_a=*/false, /*reuse_b=*/false);
}

// Load a lane's 16-bf16 fragment as two contiguous 8-element (16B) chunks.
static __device__ inline v16bf frag2(const __bf16* p0, const __bf16* p1) {
    v8bf lo = *(const v8bf*)p0;
    v8bf hi = *(const v8bf*)p1;
    return __builtin_shufflevector(lo, hi, 0,1,2,3,4,5,6,7,8,9,10,11,12,13,14,15);
}

// out[n*K + k] = bf16(in[k*N + n])  — transpose + downconvert
__global__ void convtr_kernel(const float* __restrict__ in, __bf16* __restrict__ out,
                              int K, int N) {
    int i = blockIdx.x * blockDim.x + threadIdx.x;
    if (i < K * N) {
        int n = i / K, k = i % K;
        out[i] = (__bf16)in[k * N + n];
    }
}

__global__ __launch_bounds__(512, 1)
void rnn_persist_kernel(const float* __restrict__ u, const float* __restrict__ h0,
                        const __bf16* __restrict__ WhhT, const __bf16* __restrict__ WuhT,
                        const __bf16* __restrict__ WhyT,
                        const float* __restrict__ bh, const float* __restrict__ by,
                        float* __restrict__ out)
{
    extern __shared__ __align__(16) char smem[];
    float*  hs = (float*)(smem + LDS_HS);      // [16][NH]
    float*  us = (float*)(smem + LDS_USTAGE);  // [16][NIN]
    __bf16* hb = (__bf16*)(smem + LDS_HB);     // [16][NH]
    __bf16* ub = (__bf16*)(smem + LDS_UB);     // [16][NIN]

    const int tid  = threadIdx.x;
    const int wave = tid >> 5;
    const int lane = tid & 31;
    const int nlo  = lane & 15;     // N (or M) index within tile
    const int hiH  = lane >> 4;     // half-wave selector
    const int m0   = blockIdx.x * 16;  // batch-tile base row

    // A-operand: lanes 0-15 take K 0-15, lanes 16-31 take K 16-31 (contiguous)
    const int aoff = hiH ? 16 : 0;
    // B-operand: lanes 0-15 take K 0-15, lanes 16-31 take K 16-31 (contiguous)
    const int boff = hiH ? 16 : 0;

    // ---- init hidden state from h0 ----
    for (int i = tid; i < 16 * RNN_NH; i += 512) {
        int r = i >> 9, k = i & (RNN_NH - 1);
        float v = h0[(size_t)(m0 + r) * RNN_NH + k];
        hs[i] = v;
        hb[i] = (__bf16)v;
    }
    __syncthreads();

    const float alpha = RNN_ALPHA, oma = 1.0f - RNN_ALPHA;

    #pragma clang loop unroll(disable)
    for (int t = 0; t < RNN_T; ++t) {
        // ---- (1) TDM: async copy u_t batch-tile [16 rows x 128 f32] -> LDS ----
#if __has_builtin(__builtin_amdgcn_tensor_load_to_lds)
        if (wave == 0) {
            uint64_t ga = (uint64_t)(uintptr_t)u
                        + ((uint64_t)m0 * RNN_T + (uint64_t)t) * RNN_NIN * 4u;
            v4u g0;
            g0[0] = 1u;                                  // count=1 (valid user D#)
            g0[1] = LDS_USTAGE;                          // lds_addr
            g0[2] = (unsigned)ga;                        // global_addr[31:0]
            g0[3] = (unsigned)(ga >> 32) | (2u << 30);   // global_addr[56:32] | type=2
            v8i g1;
            g1[0] = (int)(2u << 16);                     // wg_mask=0, data_size=2 (4B)
            g1[1] = (int)((RNN_NIN & 0xFFFF) << 16);     // tensor_dim0 = 128
            g1[2] = (int)((16u & 0xFFFF) << 16);         // tensor_dim1 = 16 rows
            g1[3] = (int)((RNN_NIN & 0xFFFF) << 16);     // tile_dim0 = 128
            g1[4] = 16;                                  // tile_dim1 = 16, tile_dim2 = 0
            g1[5] = (int)(RNN_T * RNN_NIN);              // tensor_dim0_stride = T*NIN
            g1[6] = 0;
            g1[7] = 0;
            v4i gz4 = {0, 0, 0, 0};
            v8i gz8 = {0, 0, 0, 0, 0, 0, 0, 0};
            __builtin_amdgcn_tensor_load_to_lds(g0, g1, gz4, gz4, gz8, 0);
            __builtin_amdgcn_s_wait_tensorcnt(0);
        }
#else
        for (int i = tid; i < 16 * RNN_NIN; i += 512) {
            int r = i >> 7, k = i & (RNN_NIN - 1);
            us[i] = u[((size_t)(m0 + r) * RNN_T + t) * RNN_NIN + k];
        }
#endif

        // ---- (2) pre = h @ W_hh : each wave owns two 16-col tiles ----
        const int n0a = wave * 32, n0b = n0a + 16;
        v8f acc0 = {}, acc1 = {};
        #pragma clang loop unroll_count(2)
        for (int k0 = 0; k0 < RNN_NH; k0 += 32) {
            const __bf16* ap = hb + nlo * RNN_NH + k0 + aoff;
            v16bf A = frag2(ap, ap + 8);
            const __bf16* bp0 = WhhT + (size_t)(n0a + nlo) * RNN_NH + k0 + boff;
            const __bf16* bp1 = WhhT + (size_t)(n0b + nlo) * RNN_NH + k0 + boff;
            v16bf B0 = frag2(bp0, bp0 + 8);
            v16bf B1 = frag2(bp1, bp1 + 8);
            acc0 = wmma_bf16(A, B0, acc0);
            acc1 = wmma_bf16(A, B1, acc1);
        }
        __syncthreads();   // u_t staged (wave0 waited TENSORcnt) + all hb reads done

        // ---- (3) convert u_t to bf16 ----
        for (int i = tid; i < 16 * RNN_NIN; i += 512)
            ub[i] = (__bf16)us[i];
        __syncthreads();

        // ---- (4) pre += u_t @ W_uh ----
        #pragma clang loop unroll_count(2)
        for (int k0 = 0; k0 < RNN_NIN; k0 += 32) {
            const __bf16* ap = ub + nlo * RNN_NIN + k0 + aoff;
            v16bf A = frag2(ap, ap + 8);
            const __bf16* bp0 = WuhT + (size_t)(n0a + nlo) * RNN_NIN + k0 + boff;
            const __bf16* bp1 = WuhT + (size_t)(n0b + nlo) * RNN_NIN + k0 + boff;
            acc0 = wmma_bf16(A, frag2(bp0, bp0 + 8), acc0);
            acc1 = wmma_bf16(A, frag2(bp1, bp1 + 8), acc1);
        }

        // ---- (5) bias + leaky tanh update; C/D layout: VGPR r -> M=r(+8), N=lane&15
        {
            float bh0 = bh[n0a + nlo], bh1 = bh[n0b + nlo];
            #pragma unroll
            for (int r = 0; r < 8; ++r) {
                int M = r + hiH * 8;
                float p0 = acc0[r] + bh0;
                float p1 = acc1[r] + bh1;
                int i0 = M * RNN_NH + n0a + nlo;
                int i1 = M * RNN_NH + n0b + nlo;
                float hn0 = oma * hs[i0] + alpha * tanhf(p0);
                float hn1 = oma * hs[i1] + alpha * tanhf(p1);
                hs[i0] = hn0; hb[i0] = (__bf16)hn0;
                hs[i1] = hn1; hb[i1] = (__bf16)hn1;
            }
        }
        __syncthreads();   // new h visible to everyone

        // ---- (6) y_t = h @ W_hy + b_y : waves 0..7, one 16-col tile each ----
        if (wave < 8) {
            const int n0 = wave * 16;
            float byv = by[n0 + nlo];
            v8f acc;
            #pragma unroll
            for (int r = 0; r < 8; ++r) acc[r] = byv;
            #pragma clang loop unroll_count(2)
            for (int k0 = 0; k0 < RNN_NH; k0 += 32) {
                const __bf16* ap = hb + nlo * RNN_NH + k0 + aoff;
                v16bf A = frag2(ap, ap + 8);
                const __bf16* bp = WhyT + (size_t)(n0 + nlo) * RNN_NH + k0 + boff;
                acc = wmma_bf16(A, frag2(bp, bp + 8), acc);
            }
            #pragma unroll
            for (int r = 0; r < 8; ++r) {
                int M = r + hiH * 8;
                out[((size_t)(m0 + M) * RNN_T + t) * RNN_NOUT + n0 + nlo] = acc[r];
            }
        }
        __syncthreads();   // phase-2 hb reads done before next step rewrites hb
    }
}

extern "C" void kernel_launch(void* const* d_in, const int* in_sizes, int n_in,
                              void* d_out, int out_size, void* d_ws, size_t ws_size,
                              hipStream_t stream) {
    (void)in_sizes; (void)n_in; (void)out_size; (void)ws_size;
    const float* u   = (const float*)d_in[0];   // [B, T, NIN]
    const float* h0  = (const float*)d_in[1];   // [B, NH]
    const float* Wuh = (const float*)d_in[2];   // [NIN, NH]
    const float* Whh = (const float*)d_in[3];   // [NH, NH]
    const float* Why = (const float*)d_in[4];   // [NH, NOUT]
    const float* bh  = (const float*)d_in[5];   // [NH]
    const float* by  = (const float*)d_in[6];   // [NOUT]
    float* out = (float*)d_out;                 // [B, T, NOUT]

    __bf16* WhhT = (__bf16*)((char*)d_ws + WS_WHHT);
    __bf16* WuhT = (__bf16*)((char*)d_ws + WS_WUHT);
    __bf16* WhyT = (__bf16*)((char*)d_ws + WS_WHYT);

    const int thr = 256;
    // W_hh: K=NH rows, N=NH cols  -> WhhT[n*NH + k]
    convtr_kernel<<<(RNN_NH * RNN_NH + thr - 1) / thr, thr, 0, stream>>>(Whh, WhhT, RNN_NH, RNN_NH);
    // W_uh: K=NIN, N=NH           -> WuhT[n*NIN + k]
    convtr_kernel<<<(RNN_NIN * RNN_NH + thr - 1) / thr, thr, 0, stream>>>(Wuh, WuhT, RNN_NIN, RNN_NH);
    // W_hy: K=NH, N=NOUT          -> WhyT[n*NH + k]
    convtr_kernel<<<(RNN_NH * RNN_NOUT + thr - 1) / thr, thr, 0, stream>>>(Why, WhyT, RNN_NH, RNN_NOUT);

    // 4 persistent workgroups (one 16-row batch tile each), 16 waves of 32
    rnn_persist_kernel<<<4, 512, LDS_BYTES, stream>>>(u, h0, WhhT, WuhT, WhyT, bh, by, out);
}
